// LGNNCore_20366734917673
// MI455X (gfx1250) — matrix operand order repaired and checked by reading, not compiled, not run on hardware
//
#include <hip/hip_runtime.h>
#include <cstdint>
#include <cstddef>

#define NN 8192
#define MM 16384
#define EE 131072
#define FD 128

typedef __attribute__((ext_vector_type(16))) __bf16 v16bf;
typedef __attribute__((ext_vector_type(8)))  __bf16 v8bf;
typedef __attribute__((ext_vector_type(8)))  float  v8f;

// ---------------------------------------------------------------- zero fill
__global__ void zero_kernel(float* __restrict__ p, int n) {
    int i = blockIdx.x * blockDim.x + threadIdx.x;
    int stride = gridDim.x * blockDim.x;
    for (; i < n; i += stride) p[i] = 0.0f;
}

// --------------------------------------- weight transpose+convert: WT[c][k]
__global__ void wt_kernel(const float* __restrict__ W, __bf16* __restrict__ WT) {
    int idx = blockIdx.x * blockDim.x + threadIdx.x;   // 16384 threads
    int k = idx >> 7, col = idx & (FD - 1);
    WT[(size_t)col * FD + k] = (__bf16)W[(size_t)k * FD + col];
}

// -------------------------------------------------- edge-list segment_sum
__global__ void agg_kernel(const float* __restrict__ zin,
                           const int* __restrict__ src,
                           const int* __restrict__ dst,
                           float* __restrict__ zout) {
    int tid  = blockIdx.x * blockDim.x + threadIdx.x;
    int e    = tid >> 5;           // wave-uniform
    int lane = tid & 31;
    if (e >= EE) return;
    int s = src[e];
    int d = dst[e];
    const float4 v = *(const float4*)(zin + (size_t)s * FD + lane * 4);
    float* o = zout + (size_t)d * FD + lane * 4;
    unsafeAtomicAdd(o + 0, v.x);
    unsafeAtomicAdd(o + 1, v.y);
    unsafeAtomicAdd(o + 2, v.z);
    unsafeAtomicAdd(o + 3, v.w);
}

// ---------------------------------------------------------- WMMA helpers
// A fragment (16x32 bf16): lane-half selects kb (+0 / +8); element i<8 -> K=kb+i,
// i>=8 -> K=kb+16+i-8 (ISA 7.12.2 16-bit A layout).
__device__ __forceinline__ v16bf a_frag_f32(const float* __restrict__ arow,
                                            int kb, float sc) {
    v16bf a;
#pragma unroll
    for (int i = 0; i < 8; ++i) {
        a[i]     = (__bf16)(arow[kb + i] * sc);
        a[i + 8] = (__bf16)(arow[kb + 16 + i] * sc);
    }
    return a;
}

// X[16x128] @ W[128x16] -> acc, with W pre-transposed bf16 (WT[col][k]).
__device__ __forceinline__ void gemm_k128(const float* __restrict__ arow,
                                          const __bf16* __restrict__ WT,
                                          float sc, int col, int lane, v8f& acc) {
    int kb_off  = (lane >> 4) << 3;
    int kbB_off = (lane >> 4) << 4;     // unused: B loads use kb_off, same K order as A
    const __bf16* brow = WT + (size_t)col * FD;
#pragma unroll
    for (int k0 = 0; k0 < FD; k0 += 32) {
        v16bf a = a_frag_f32(arow, k0 + kb_off, sc);
        v16bf b = __builtin_shufflevector(*(const v8bf*)(brow + k0 + kb_off), *(const v8bf*)(brow + k0 + kb_off + 16), 0, 1, 2, 3, 4, 5, 6, 7, 8, 9, 10, 11, 12, 13, 14, 15);   // two 16B halves: K=kb..kb+7, kb+16..kb+23
        acc = __builtin_amdgcn_wmma_f32_16x16x32_bf16(
            false, a, false, b, (short)0, acc, false, false);
    }
}

// ---------------------------------------------- fbT = (feat_b @ W_half)^T
__global__ __launch_bounds__(256)
void fbt_gemm_kernel(const float* __restrict__ feat_b,
                     const __bf16* __restrict__ WTf0,
                     const __bf16* __restrict__ WTf1,
                     __bf16* __restrict__ fbT0,
                     __bf16* __restrict__ fbT1) {
    int lane = threadIdx.x & 31;
    int wv   = threadIdx.x >> 5;
    int m0   = blockIdx.x * 16;
    int col0 = wv * 16;
    int row  = m0 + (lane & 15);
    int col  = col0 + (lane & 15);
    const float* arow = feat_b + (size_t)row * FD;
    int hi8 = (lane >> 4) << 3;
#pragma unroll
    for (int ch = 0; ch < 2; ++ch) {
        v8f acc = {};
        gemm_k128(arow, ch ? WTf1 : WTf0, 1.0f, col, lane, acc);
        __bf16* outp = (ch ? fbT1 : fbT0) + (size_t)col * MM + m0 + hi8;
        v8bf t;
#pragma unroll
        for (int r = 0; r < 8; ++r) t[r] = (__bf16)acc[r];
        *(v8bf*)outp = t;   // 16B contiguous (m contiguous per lane-half)
    }
}

// ------------------------------------- fuse = pm0 @ fbT0^T + pm1 @ fbT1^T
// LDS-staged, double-buffered, software-pipelined. One float2 = both channels.
#define KSTEP 64
// [k][row] with k-group pad so k and k+8 hit disjoint bank halves
#define LIDX(k, r) ((((k) + ((k) >> 3)) << 4) + (r))

__global__ __launch_bounds__(256)
void fuse_gemm_kernel(const float* __restrict__ pm_pd,
                      const __bf16* __restrict__ fbT0,
                      const __bf16* __restrict__ fbT1,
                      float* __restrict__ fuse) {
    __shared__ float2 abuf[2][1152];          // 2 x 9KB
    int tid  = threadIdx.x;
    int lane = tid & 31;
    int wv   = tid >> 5;
    int row0 = blockIdx.x * 16;
    int col  = wv * 16 + (lane & 15);
    int lrc  = lane & 15;

    // cooperative-load mapping: lr = row, k = lk + 16*j (strided k for banks)
    int lr = tid & 15;
    int lk = tid >> 4;                        // 0..15
    const float2* gsrc = (const float2*)pm_pd + (size_t)(row0 + lr) * MM;

    const __bf16* b0base = fbT0 + (size_t)col * MM;
    const __bf16* b1base = fbT1 + (size_t)col * MM;
    int kb_off  = (lane >> 4) << 3;
    int kbB_off = (lane >> 4) << 4;

    // prologue: block 0 -> buf 0
    float2 rg[4];
#pragma unroll
    for (int j = 0; j < 4; ++j) rg[j] = gsrc[lk + 16 * j];
#pragma unroll
    for (int j = 0; j < 4; ++j) abuf[0][LIDX(lk + 16 * j, lr)] = rg[j];

    v8f acc = {};
    int p = 0;
    for (int k0 = 0; k0 < MM; k0 += KSTEP) {
        __syncthreads();                      // stores of block k0 visible
        bool more = (k0 + KSTEP) < MM;
        if (more) {                           // issue next block's global loads
            const float2* g = gsrc + (k0 + KSTEP);
#pragma unroll
            for (int j = 0; j < 4; ++j) rg[j] = g[lk + 16 * j];
            __builtin_prefetch(gsrc + (k0 + 2 * KSTEP), 0, 0);
        }
#pragma unroll
        for (int s = 0; s < 2; ++s) {         // two 32-wide sub-blocks
            int kb = s * 32 + kb_off;
            v16bf a0, a1;
#pragma unroll
            for (int i = 0; i < 8; ++i) {
                float2 pv = abuf[p][LIDX(kb + i, lrc)];
                float2 qv = abuf[p][LIDX(kb + 16 + i, lrc)];
                a0[i]     = (__bf16)pv.x;  a1[i]     = (__bf16)pv.y;
                a0[i + 8] = (__bf16)qv.x;  a1[i + 8] = (__bf16)qv.y;
            }
            v16bf b0 = __builtin_shufflevector(*(const v8bf*)(b0base + k0 + kb), *(const v8bf*)(b0base + k0 + kb + 16), 0, 1, 2, 3, 4, 5, 6, 7, 8, 9, 10, 11, 12, 13, 14, 15);
            v16bf b1 = __builtin_shufflevector(*(const v8bf*)(b1base + k0 + kb), *(const v8bf*)(b1base + k0 + kb + 16), 0, 1, 2, 3, 4, 5, 6, 7, 8, 9, 10, 11, 12, 13, 14, 15);
            acc = __builtin_amdgcn_wmma_f32_16x16x32_bf16(
                false, a0, false, b0, (short)0, acc, false, false);
            acc = __builtin_amdgcn_wmma_f32_16x16x32_bf16(
                false, a1, false, b1, (short)0, acc, false, false);
        }
        if (more) {                           // stage next block into other buf
#pragma unroll
            for (int j = 0; j < 4; ++j) abuf[p ^ 1][LIDX(lk + 16 * j, lr)] = rg[j];
        }
        p ^= 1;
    }

    int hi8 = (lane >> 4) << 3;
#pragma unroll
    for (int r = 0; r < 8; ++r)
        fuse[(size_t)(row0 + r + hi8) * FD + col] = acc[r];
}

// --------- proj = feat_a@Wp + (deg*feat_a)@Wd + z1@Wr0 + z2@Wr1 + z4@Wr2
__global__ __launch_bounds__(256)
void proj_gemm_kernel(const float* __restrict__ feat_a,
                      const float* __restrict__ deg,
                      const float* __restrict__ z1,
                      const float* __restrict__ z2,
                      const float* __restrict__ z4,
                      const __bf16* __restrict__ WT,   // 5 mats, 16384 each
                      float* __restrict__ proj) {
    int lane = threadIdx.x & 31;
    int wv   = threadIdx.x >> 5;
    int row0 = blockIdx.x * 16;
    int col  = wv * 16 + (lane & 15);
    int row  = row0 + (lane & 15);
    float dsc = deg[row];
    v8f acc = {};
    gemm_k128(feat_a + (size_t)row * FD, WT,              1.0f, col, lane, acc);
    gemm_k128(feat_a + (size_t)row * FD, WT + 1 * FD * FD, dsc, col, lane, acc);
    gemm_k128(z1     + (size_t)row * FD, WT + 2 * FD * FD, 1.0f, col, lane, acc);
    gemm_k128(z2     + (size_t)row * FD, WT + 3 * FD * FD, 1.0f, col, lane, acc);
    gemm_k128(z4     + (size_t)row * FD, WT + 4 * FD * FD, 1.0f, col, lane, acc);
    int hi8 = (lane >> 4) << 3;
#pragma unroll
    for (int r = 0; r < 8; ++r)
        proj[(size_t)(row0 + r + hi8) * FD + col] = acc[r];
}

// -------- combine: result = proj + fuse + biases, half-ReLU, column stats
__global__ void combine_kernel(const float* __restrict__ proj,
                               const float* __restrict__ fuse,
                               const float* __restrict__ b_prev,
                               const float* __restrict__ b_deg,
                               const float* __restrict__ b_r,
                               const float* __restrict__ b_fuse,
                               float* __restrict__ out,
                               float* __restrict__ stats) {
    int c = threadIdx.x;  // 128 columns
    float bias = b_prev[c] + b_deg[c] + b_r[c] + b_r[FD + c] + b_r[2 * FD + c]
               + b_fuse[c];
    float s = 0.0f, sq = 0.0f;
    int n0 = blockIdx.x * 64;
    for (int r = 0; r < 64; ++r) {
        size_t idx = (size_t)(n0 + r) * FD + c;
        float v = proj[idx] + fuse[idx] + bias;
        if (c >= FD / 2) v = fmaxf(v, 0.0f);
        out[idx] = v;
        s += v;
        sq += v * v;
    }
    unsafeAtomicAdd(&stats[c], s);
    unsafeAtomicAdd(&stats[FD + c], sq);
}

// ---------------------------- BatchNorm (training mode, biased var), inplace
__global__ void bn_kernel(float* __restrict__ out,
                          const float* __restrict__ stats,
                          const float* __restrict__ gamma,
                          const float* __restrict__ beta) {
    size_t idx = (size_t)blockIdx.x * blockDim.x + threadIdx.x;
    int c = (int)(idx & (FD - 1));
    float mean = stats[c] * (1.0f / NN);
    float var  = stats[FD + c] * (1.0f / NN) - mean * mean;
    out[idx] = (out[idx] - mean) * rsqrtf(var + 1e-5f) * gamma[c] + beta[c];
}

// ----------------------------------------------------------------- launch
extern "C" void kernel_launch(void* const* d_in, const int* in_sizes, int n_in,
                              void* d_out, int out_size, void* d_ws, size_t ws_size,
                              hipStream_t stream) {
    const float* feat_a = (const float*)d_in[0];
    const float* feat_b = (const float*)d_in[1];
    const float* deg    = (const float*)d_in[2];
    const float* pm_pd  = (const float*)d_in[3];
    const int*   src    = (const int*)d_in[4];
    const int*   dst    = (const int*)d_in[5];
    const float* W_prev = (const float*)d_in[6];
    const float* b_prev = (const float*)d_in[7];
    const float* W_deg  = (const float*)d_in[8];
    const float* b_deg  = (const float*)d_in[9];
    const float* W_r    = (const float*)d_in[10];
    const float* b_r    = (const float*)d_in[11];
    const float* W_fuse = (const float*)d_in[12];
    const float* b_fuse = (const float*)d_in[13];
    const float* gamma  = (const float*)d_in[14];
    const float* beta   = (const float*)d_in[15];
    float* out = (float*)d_out;

    // workspace carve-up
    const size_t ZSZ = (size_t)NN * FD;          // 1 Mi floats per z buffer
    const size_t WSZ = (size_t)FD * FD;          // 16384 elems per weight
    char*  ws    = (char*)d_ws;
    float* z1    = (float*)ws;
    float* z2    = z1 + ZSZ;
    float* z3    = z2 + ZSZ;
    float* z4    = z3 + ZSZ;
    float* stats = z4 + ZSZ;                     // 256 used, 1024 padded
    __bf16* WT   = (__bf16*)(stats + 1024);      // 7 x [128][128] bf16
    __bf16* fbT0 = WT + 7 * WSZ;                 // [128][16384] bf16
    __bf16* fbT1 = fbT0 + (size_t)FD * MM;
    float* proj  = (float*)(fbT1 + (size_t)FD * MM);
    float* fuse  = proj + ZSZ;

    // 1) zero z1..z4 + stats
    int nz = (int)(4 * ZSZ + 1024);
    zero_kernel<<<2048, 256, 0, stream>>>(z1, nz);

    // 1b) weight transpose+bf16: [Wp, Wd, Wr0, Wr1, Wr2, Wf0, Wf1]
    wt_kernel<<<64, 256, 0, stream>>>(W_prev,        WT + 0 * WSZ);
    wt_kernel<<<64, 256, 0, stream>>>(W_deg,         WT + 1 * WSZ);
    wt_kernel<<<64, 256, 0, stream>>>(W_r,           WT + 2 * WSZ);
    wt_kernel<<<64, 256, 0, stream>>>(W_r + WSZ,     WT + 3 * WSZ);
    wt_kernel<<<64, 256, 0, stream>>>(W_r + 2 * WSZ, WT + 4 * WSZ);
    wt_kernel<<<64, 256, 0, stream>>>(W_fuse,        WT + 5 * WSZ);
    wt_kernel<<<64, 256, 0, stream>>>(W_fuse + WSZ,  WT + 6 * WSZ);

    // 2) message passing: z1=A x, z2=A^2 x, z3=A^3 x, z4=A^4 x
    dim3 gAgg((EE * 32) / 256);
    agg_kernel<<<gAgg, 256, 0, stream>>>(feat_a, src, dst, z1);
    agg_kernel<<<gAgg, 256, 0, stream>>>(z1,     src, dst, z2);
    agg_kernel<<<gAgg, 256, 0, stream>>>(z2,     src, dst, z3);
    agg_kernel<<<gAgg, 256, 0, stream>>>(z3,     src, dst, z4);

    // 3) fbT_c = (feat_b @ W_fuse_half_c)^T, bf16
    fbt_gemm_kernel<<<MM / 16, 256, 0, stream>>>(
        feat_b, WT + 5 * WSZ, WT + 6 * WSZ, fbT0, fbT1);

    // 4) big GEMM over pm_pd (HBM-bound, LDS-staged bf16 WMMA)
    fuse_gemm_kernel<<<NN / 16, 256, 0, stream>>>(pm_pd, fbT0, fbT1, fuse);

    // 5) all five node projections fused
    proj_gemm_kernel<<<NN / 16, 256, 0, stream>>>(
        feat_a, deg, z1, z2, z4, WT, proj);

    // 6) combine + biases + half-ReLU + column stats
    combine_kernel<<<NN / 64, FD, 0, stream>>>(
        proj, fuse, b_prev, b_deg, b_r, b_fuse, out, stats);

    // 7) batchnorm in place
    bn_kernel<<<(NN * FD) / 256, 256, 0, stream>>>(out, stats, gamma, beta);
}
